// rnn_88965952569377
// MI455X (gfx1250) — compile-verified, matrix operand order checked by reading, not run on hardware
//
#include <hip/hip_runtime.h>
#include <math.h>

typedef __attribute__((ext_vector_type(16))) __bf16       v16bf;
typedef __attribute__((ext_vector_type(8)))  float        v8f;
typedef __attribute__((ext_vector_type(8)))  unsigned int v8u;
typedef unsigned int u32x4 __attribute__((ext_vector_type(4)));
typedef int          i32x8 __attribute__((ext_vector_type(8)));
typedef int          i32x4 __attribute__((ext_vector_type(4)));
typedef unsigned short u16;

#if defined(__HIP_DEVICE_COMPILE__) && __has_builtin(__builtin_amdgcn_tensor_load_to_lds) && __has_builtin(__builtin_amdgcn_s_wait_tensorcnt)
#define USE_TDM 1
#else
#define USE_TDM 0
#endif

// ---------- bf16 helpers (storage = ushort, manual RNE convert) ----------
__device__ __forceinline__ float bf2f(u16 h) {
  return __uint_as_float(((unsigned)h) << 16);
}
__device__ __forceinline__ u16 f2bf(float f) {
  unsigned u = __float_as_uint(f);
  u += 0x7FFFu + ((u >> 16) & 1u);
  return (u16)(u >> 16);
}

// Load a 16x32 bf16 WMMA operand fragment (A layout per CDNA5 ISA 7.12.2):
// lane L: row = L&15, k-half = L>>4; dword i holds K pair
// k = (i>=4 ? 16 : 0) + (L>>4)*8 + (i&3)*2. W[N,K] row-major reuses this as
// the transposed (B) operand with N in place of M. Also valid on LDS pointers
// (lowers to ds_load_b128 after merging).
template <typename P>
__device__ __forceinline__ v16bf load_frag(P base, int ld, int k0, int lane) {
  P row = base + (size_t)(lane & 15) * ld + k0 + ((lane >> 4) << 3);
  v8u raw;
#pragma unroll
  for (int i = 0; i < 8; ++i) {
    int k = ((i & 4) << 2) + ((i & 3) << 1);
    raw[i] = *(const unsigned int*)(row + k);
  }
  return __builtin_bit_cast(v16bf, raw);
}

// ---------- Tensor Data Mover: 2D bf16 tile Global -> LDS ----------
// D# per CDNA5 ISA 8.3/8.4: group0 = {count, lds_addr, global_addr, type=2},
// group1 = {data_size=2B, tensor dims, tile dims, dim0 stride}.
__device__ __forceinline__ void tdm_load_2d_bf16(
    unsigned lds_off, const void* gaddr, unsigned tensor_d0, unsigned tensor_d1,
    unsigned tile_d0, unsigned tile_d1, unsigned stride0) {
#if USE_TDM
  unsigned long long ga = (unsigned long long)gaddr;
  u32x4 g0;
  g0[0] = 1u;                                   // count=1, user descriptor
  g0[1] = lds_off;                              // lds_addr (bytes)
  g0[2] = (unsigned)(ga & 0xFFFFFFFFu);         // global_addr[31:0]
  g0[3] = (unsigned)((ga >> 32) & 0x01FFFFFFu)  // global_addr[56:32]
          | 0x80000000u;                        // type=2 ("image")
  i32x8 g1;
  g1[0] = (int)(1u << 16);                      // data_size=1 -> 2 bytes
  g1[1] = (int)((tensor_d0 & 0xFFFFu) << 16);   // tensor_dim0[15:0]
  g1[2] = (int)(((tensor_d0 >> 16) & 0xFFFFu) | ((tensor_d1 & 0xFFFFu) << 16));
  g1[3] = (int)(((tensor_d1 >> 16) & 0xFFFFu) | ((tile_d0 & 0xFFFFu) << 16));
  g1[4] = (int)(tile_d1 & 0xFFFFu);             // tile_dim1, tile_dim2=0
  g1[5] = (int)stride0;                         // tensor_dim0_stride[31:0]
  g1[6] = 0;
  g1[7] = 0;
  i32x4 z4 = {0, 0, 0, 0};
#if __clang_major__ >= 23
  i32x8 z8 = {0, 0, 0, 0, 0, 0, 0, 0};
  __builtin_amdgcn_tensor_load_to_lds(g0, g1, z4, z4, z8, 0);
#else
  __builtin_amdgcn_tensor_load_to_lds(g0, g1, z4, z4, 0);
#endif
#else
  (void)lds_off; (void)gaddr; (void)tensor_d0; (void)tensor_d1;
  (void)tile_d0; (void)tile_d1; (void)stride0;
#endif
}

// ---------- generic bf16 WMMA GEMM: C[M,N] = A[M,K] * W[N,K]^T + bias ----------
template <int MT, int NT, bool OUTBF>
__global__ __launch_bounds__(256) void k_gemm(
    const u16* __restrict__ A, const u16* __restrict__ W,
    const float* __restrict__ bias, void* __restrict__ Cout,
    int M, int N, int K, int lda, int ldw, int ldc) {
  const int lane = threadIdx.x & 31;
  const int wave = blockIdx.x * 8 + (threadIdx.x >> 5);
  const int ntw  = N / (NT * 16);
  const int wm   = wave / ntw;
  if (wm * MT * 16 >= M) return;
  const int m0 = wm * MT * 16;
  const int n0 = (wave % ntw) * NT * 16;

  const v8f vzero = {0.f, 0.f, 0.f, 0.f, 0.f, 0.f, 0.f, 0.f};
  v8f acc[MT][NT];
#pragma unroll
  for (int i = 0; i < MT; ++i)
#pragma unroll
    for (int j = 0; j < NT; ++j) acc[i][j] = vzero;

  for (int k0 = 0; k0 < K; k0 += 32) {
    if (k0 + 32 < K) {  // CDNA5 global_prefetch_b8 for next K slab
      __builtin_prefetch(A + (size_t)(m0 + (lane & 15)) * lda + k0 + 32, 0, 3);
      __builtin_prefetch(W + (size_t)(n0 + (lane & 15)) * ldw + k0 + 32, 0, 3);
    }
    v16bf af[MT], bfrag[NT];
#pragma unroll
    for (int i = 0; i < MT; ++i)
      af[i] = load_frag(A + (size_t)(m0 + i * 16) * lda, lda, k0, lane);
#pragma unroll
    for (int j = 0; j < NT; ++j)
      bfrag[j] = load_frag(W + (size_t)(n0 + j * 16) * ldw, ldw, k0, lane);
#pragma unroll
    for (int i = 0; i < MT; ++i)
#pragma unroll
      for (int j = 0; j < NT; ++j)
        acc[i][j] = __builtin_amdgcn_wmma_f32_16x16x32_bf16(
            false, af[i], false, bfrag[j], (short)0, acc[i][j], false, false);
  }

  // C/D layout: VGPR v, lanes 0-15 -> row v, lanes 16-31 -> row v+8; col = lane&15.
  const int rb = (lane >> 4) * 8;
  const int cl = lane & 15;
#pragma unroll
  for (int i = 0; i < MT; ++i)
#pragma unroll
    for (int j = 0; j < NT; ++j) {
      const int col = n0 + j * 16 + cl;
      const float bv = bias ? bias[col] : 0.0f;
#pragma unroll
      for (int v = 0; v < 8; ++v) {
        const int row = m0 + i * 16 + rb + v;
        const float val = acc[i][j][v] + bv;
        if (OUTBF)
          ((u16*)Cout)[(size_t)row * ldc + col] = f2bf(val);
        else
          ((float*)Cout)[(size_t)row * ldc + col] = val;
      }
    }
}

// ---------- fused GRU step: gh = h @ W_hh^T + b_hh, then gate math ----------
// Grid: 8 blocks x 256. Block b: wm = b>>2 (rows wm*32..+31), waves own one
// 16-col slice jt = (b&3)*8 + wave of the hidden state and all 3 gate
// sections (N offsets jt*16, 512+jt*16, 1024+jt*16).
// A slab (32x512 bf16, 32KB) staged into LDS once via TDM, then read as
// ds_load fragments each K step.
__global__ __launch_bounds__(256) void k_step(
    const u16* __restrict__ Ain,   // hbf (previous hidden, bf16)
    const u16* __restrict__ Whh, const float* __restrict__ bhh,
    const u16* __restrict__ gi, float* __restrict__ hf,
    u16* __restrict__ Aout,        // new hidden bf16 (double buffer)
    u16* __restrict__ hseq, int t) {
  __shared__ __align__(16) u16 Atile[32 * 512];
  const int lane = threadIdx.x & 31;
  const int wave = threadIdx.x >> 5;
  const int wm   = blockIdx.x >> 2;
  const int jt   = (blockIdx.x & 3) * 8 + wave;
  const int m0   = wm * 32;

#if USE_TDM
  if (threadIdx.x == 0)
    tdm_load_2d_bf16((unsigned)(size_t)(&Atile[0]), Ain + (size_t)m0 * 512,
                     /*tensor_d0=*/512, /*tensor_d1=*/64,
                     /*tile_d0=*/512, /*tile_d1=*/32, /*stride0=*/512);
  if (threadIdx.x < 32) __builtin_amdgcn_s_wait_tensorcnt(0);
  __syncthreads();
#else
  {
    const u16* src = Ain + (size_t)m0 * 512;
    for (int i = threadIdx.x * 8; i < 32 * 512; i += 256 * 8)
      *(uint4*)&Atile[i] = *(const uint4*)&src[i];
    __syncthreads();
  }
#endif

  const v8f vzero = {0.f, 0.f, 0.f, 0.f, 0.f, 0.f, 0.f, 0.f};
  v8f acc[3][2];  // [gate section r/z/n][row tile]
#pragma unroll
  for (int s = 0; s < 3; ++s)
#pragma unroll
    for (int i = 0; i < 2; ++i) acc[s][i] = vzero;

  for (int k0 = 0; k0 < 512; k0 += 32) {
    v16bf af[2], bfrag[3];
#pragma unroll
    for (int i = 0; i < 2; ++i)
      af[i] = load_frag(&Atile[(size_t)(i * 16) * 512], 512, k0, lane);
#pragma unroll
    for (int s = 0; s < 3; ++s)
      bfrag[s] = load_frag(Whh + (size_t)(s * 512 + jt * 16) * 512, 512, k0, lane);
#pragma unroll
    for (int s = 0; s < 3; ++s)
#pragma unroll
      for (int i = 0; i < 2; ++i)
        acc[s][i] = __builtin_amdgcn_wmma_f32_16x16x32_bf16(
            false, af[i], false, bfrag[s], (short)0, acc[s][i], false, false);
  }

  const int rb = (lane >> 4) * 8;
  const int cl = lane & 15;
  const int col = jt * 16 + cl;
  const float br = bhh[col], bz = bhh[512 + col], bn = bhh[1024 + col];
#pragma unroll
  for (int i = 0; i < 2; ++i)
#pragma unroll
    for (int v = 0; v < 8; ++v) {
      const int row = m0 + i * 16 + rb + v;  // batch index b
      const u16* git = gi + ((size_t)row * 256 + t) * 1536;
      const float hr = acc[0][i][v] + br;
      const float hz = acc[1][i][v] + bz;
      const float hn = acc[2][i][v] + bn;
      const float ir = bf2f(git[col]);
      const float iz = bf2f(git[col + 512]);
      const float in_ = bf2f(git[col + 1024]);
      const float rg = 1.0f / (1.0f + expf(-(ir + hr)));
      const float zg = 1.0f / (1.0f + expf(-(iz + hz)));
      const float ng = tanhf(in_ + rg * hn);
      const size_t hidx = (size_t)row * 512 + col;
      const float hnew = (1.0f - zg) * ng + zg * hf[hidx];
      hf[hidx] = hnew;
      const u16 hb = f2bf(hnew);
      Aout[hidx] = hb;
      hseq[((size_t)row * 256 + t) * 512 + col] = hb;
    }
}

// ---------- fused mu/logsig GEMMs + diagonal-Gaussian log-prob ----------
__global__ __launch_bounds__(256) void k_musig(
    const u16* __restrict__ A, const u16* __restrict__ Wm,
    const u16* __restrict__ Ws, const float* __restrict__ bm,
    const float* __restrict__ bs, const float* __restrict__ x,
    float* __restrict__ lp) {
  const int lane = threadIdx.x & 31;
  const int wave = blockIdx.x * 8 + (threadIdx.x >> 5);
  const int ntw  = 768 / 32;
  const int m0   = (wave / ntw) * 32;
  if (m0 >= 16384) return;
  const int n0 = (wave % ntw) * 32;

  const v8f vzero = {0.f, 0.f, 0.f, 0.f, 0.f, 0.f, 0.f, 0.f};
  v8f am[2][2], as_[2][2];
#pragma unroll
  for (int i = 0; i < 2; ++i)
#pragma unroll
    for (int j = 0; j < 2; ++j) { am[i][j] = vzero; as_[i][j] = vzero; }

  for (int k0 = 0; k0 < 768; k0 += 32) {
    v16bf af[2], bmf[2], bsf[2];
#pragma unroll
    for (int i = 0; i < 2; ++i)
      af[i] = load_frag(A + (size_t)(m0 + i * 16) * 768, 768, k0, lane);
#pragma unroll
    for (int j = 0; j < 2; ++j) {
      bmf[j] = load_frag(Wm + (size_t)(n0 + j * 16) * 768, 768, k0, lane);
      bsf[j] = load_frag(Ws + (size_t)(n0 + j * 16) * 768, 768, k0, lane);
    }
#pragma unroll
    for (int i = 0; i < 2; ++i)
#pragma unroll
      for (int j = 0; j < 2; ++j) {
        am[i][j] = __builtin_amdgcn_wmma_f32_16x16x32_bf16(
            false, af[i], false, bmf[j], (short)0, am[i][j], false, false);
        as_[i][j] = __builtin_amdgcn_wmma_f32_16x16x32_bf16(
            false, af[i], false, bsf[j], (short)0, as_[i][j], false, false);
      }
  }

  const int rb = (lane >> 4) * 8;
  const int cl = lane & 15;
#pragma unroll
  for (int i = 0; i < 2; ++i)
#pragma unroll
    for (int v = 0; v < 8; ++v) {
      const int row = m0 + i * 16 + rb + v;
      float contrib = 0.0f;
#pragma unroll
      for (int j = 0; j < 2; ++j) {
        const int col = n0 + j * 16 + cl;
        const float mu = am[i][j][v] + bm[col];
        const float ls = as_[i][j][v] + bs[col];
        const float d  = x[(size_t)row * 768 + col] - mu;
        contrib += d * d * expf(-ls) + ls;
      }
      // lanes (L&15)=0..15 of each 16-lane half hold the same row -> reduce
#pragma unroll
      for (int off = 1; off < 16; off <<= 1)
        contrib += __shfl_xor(contrib, off, 32);
      if (cl == 0) atomicAdd(&lp[row], contrib);
    }
}

// ---------- fp32 -> bf16 conversion with optional K zero-padding ----------
__global__ __launch_bounds__(256) void k_cvt(const float* __restrict__ src,
                                             u16* __restrict__ dst, int cols,
                                             int ldd, size_t tot) {
  size_t i = (size_t)blockIdx.x * 256 + threadIdx.x;
  if (i >= tot) return;
  size_t r = i / (size_t)ldd;
  int c = (int)(i - r * (size_t)ldd);
  dst[i] = (c < cols) ? f2bf(src[r * (size_t)cols + c]) : (u16)0;
}

// ---------- block reduction helper ----------
__device__ __forceinline__ float block_reduce_sum(float v, float* red) {
  const int tid = threadIdx.x;
  red[tid] = v;
  __syncthreads();
  for (int s = 128; s > 0; s >>= 1) {
    if (tid < s) red[tid] += red[tid + s];
    __syncthreads();
  }
  float r = red[0];
  __syncthreads();
  return r;
}

// ---------- LayerNorm + ReLU -> bf16 (zero-pads to ldout) ----------
__global__ __launch_bounds__(256) void k_ln_relu(
    const float* __restrict__ in, const float* __restrict__ g,
    const float* __restrict__ b, u16* __restrict__ out, int N, int ldin,
    int ldout) {
  __shared__ float red[256];
  const int row = blockIdx.x;
  const float* xr = in + (size_t)row * ldin;
  float s = 0.0f;
  for (int c = threadIdx.x; c < N; c += 256) s += xr[c];
  const float mean = block_reduce_sum(s, red) / (float)N;
  float v = 0.0f;
  for (int c = threadIdx.x; c < N; c += 256) {
    float d = xr[c] - mean;
    v += d * d;
  }
  const float var  = block_reduce_sum(v, red) / (float)N;
  const float rstd = rsqrtf(var + 1e-5f);
  for (int c = threadIdx.x; c < ldout; c += 256) {
    float val = 0.0f;
    if (c < N) val = fmaxf((xr[c] - mean) * rstd * g[c] + b[c], 0.0f);
    out[(size_t)row * ldout + c] = f2bf(val);
  }
}

// ---------- h0 = tanh(noise @ W_init^T + b_init) ----------
__global__ __launch_bounds__(256) void k_h0(const float* __restrict__ noise,
                                            const float* __restrict__ Wi,
                                            const float* __restrict__ bi,
                                            float* __restrict__ h,
                                            u16* __restrict__ hbf) {
  const int idx = blockIdx.x * 256 + threadIdx.x;  // 0..32767
  const int b = idx >> 9, j = idx & 511;
  const float* nr = noise + b * 100;
  const float* wr = Wi + j * 100;
  float s = bi[j];
  for (int k = 0; k < 100; ++k) s += nr[k] * wr[k];
  const float hv = tanhf(s);
  h[idx]   = hv;
  hbf[idx] = f2bf(hv);
}

__global__ void k_zero_lp(float* __restrict__ lp, int n) {
  const int i = blockIdx.x * 256 + threadIdx.x;
  if (i < n) lp[i] = 0.0f;
}

// ---------- masked mean over timesteps -> out[b] ----------
__global__ __launch_bounds__(256) void k_final(const float* __restrict__ lp,
                                               const int* __restrict__ lengths,
                                               float* __restrict__ out) {
  __shared__ float red[256];
  const int b = blockIdx.x;
  const int len = lengths[b];
  const int t = threadIdx.x;
  float v = 0.0f;
  if (t >= 1 && t < len)
    v = -0.5f * (lp[b * 256 + t] + 768.0f * logf(2.0f * (float)M_PI));
  const float s = block_reduce_sum(v, red);
  if (t == 0) out[b] = s / (float)len;
}

extern "C" void kernel_launch(void* const* d_in, const int* in_sizes, int n_in,
                              void* d_out, int out_size, void* d_ws,
                              size_t ws_size, hipStream_t stream) {
  (void)in_sizes; (void)n_in; (void)out_size; (void)ws_size;
  const float* x        = (const float*)d_in[0];
  const int*   lengths  = (const int*)d_in[1];
  const float* h0n      = (const float*)d_in[3];
  const float* W_lin_in = (const float*)d_in[4];
  const float* b_lin_in = (const float*)d_in[5];
  const float* g_ln1    = (const float*)d_in[6];
  const float* b_ln1    = (const float*)d_in[7];
  const float* W_ih     = (const float*)d_in[8];
  const float* W_hh     = (const float*)d_in[9];
  const float* b_ih     = (const float*)d_in[10];
  const float* b_hh     = (const float*)d_in[11];
  const float* W_init   = (const float*)d_in[12];
  const float* b_init   = (const float*)d_in[13];
  const float* W_lin1   = (const float*)d_in[14];
  const float* b_lin1   = (const float*)d_in[15];
  const float* g_ln2    = (const float*)d_in[16];
  const float* b_ln2    = (const float*)d_in[17];
  const float* W_mu     = (const float*)d_in[18];
  const float* b_mu     = (const float*)d_in[19];
  const float* W_sig    = (const float*)d_in[20];
  const float* b_sig    = (const float*)d_in[21];

  char* wp = (char*)d_ws;
  auto alloc = [&](size_t bytes) -> void* {
    void* p = (void*)wp;
    wp += (bytes + 255) & ~(size_t)255;
    return p;
  };
  const int M = 64 * 256;  // flattened (b, t)
  u16*   xbf   = (u16*)alloc((size_t)M * 768 * 2);
  u16*   Wli   = (u16*)alloc((size_t)400 * 768 * 2);
  u16*   Wihb  = (u16*)alloc((size_t)1536 * 416 * 2);
  u16*   Whhb  = (u16*)alloc((size_t)1536 * 512 * 2);
  u16*   Wl1b  = (u16*)alloc((size_t)768 * 512 * 2);
  u16*   Wmub  = (u16*)alloc((size_t)768 * 768 * 2);
  u16*   Wsigb = (u16*)alloc((size_t)768 * 768 * 2);
  float* tmp   = (float*)alloc((size_t)M * 768 * 4);
  u16*   rbf   = (u16*)alloc((size_t)M * 416 * 2);
  u16*   gibf  = (u16*)alloc((size_t)M * 1536 * 2);
  float* hf    = (float*)alloc((size_t)64 * 512 * 4);
  u16*   hbf0  = (u16*)alloc((size_t)64 * 512 * 2);
  u16*   hbf1  = (u16*)alloc((size_t)64 * 512 * 2);
  u16*   hseq  = (u16*)alloc((size_t)M * 512 * 2);
  u16*   hhbf  = (u16*)alloc((size_t)M * 768 * 2);
  float* lp    = (float*)alloc((size_t)M * 4);

  auto cvt = [&](const float* src, u16* dst, int rows, int cols, int ldd) {
    size_t tot = (size_t)rows * ldd;
    k_cvt<<<(unsigned)((tot + 255) / 256), 256, 0, stream>>>(src, dst, cols, ldd, tot);
  };
  cvt(x, xbf, M, 768, 768);
  cvt(W_lin_in, Wli, 400, 768, 768);
  cvt(W_ih, Wihb, 1536, 400, 416);   // K padded 400 -> 416
  cvt(W_hh, Whhb, 1536, 512, 512);
  cvt(W_lin1, Wl1b, 768, 512, 512);
  cvt(W_mu, Wmub, 768, 768, 768);
  cvt(W_sig, Wsigb, 768, 768, 768);

  // 1) r = relu(LN(x @ W_lin_in^T + b))
  k_gemm<2, 1, false><<<1600, 256, 0, stream>>>(xbf, Wli, b_lin_in, tmp,
                                                M, 400, 768, 768, 768, 400);
  k_ln_relu<<<M, 256, 0, stream>>>(tmp, g_ln1, b_ln1, rbf, 400, 400, 416);

  // 2) gi = r @ W_ih^T + b_ih  (bf16 out, padded K)
  k_gemm<2, 4, true><<<1536, 256, 0, stream>>>(rbf, Wihb, b_ih, gibf,
                                               M, 1536, 416, 416, 416, 1536);

  // 3) h0
  k_h0<<<128, 256, 0, stream>>>(h0n, W_init, b_init, hf, hbf0);

  // 4) sequential GRU scan: one fused TDM+WMMA+gate kernel per step,
  //    double-buffered bf16 hidden state (W_hh stays hot in L2)
  u16* hA = hbf0;
  u16* hB = hbf1;
  for (int t = 0; t < 256; ++t) {
    k_step<<<8, 256, 0, stream>>>(hA, Whhb, b_hh, gibf, hf, hB, hseq, t);
    u16* tmpp = hA; hA = hB; hB = tmpp;
  }

  // 5) hh = relu(LN(h_seq @ W_lin1^T + b))
  k_gemm<2, 4, false><<<768, 256, 0, stream>>>(hseq, Wl1b, b_lin1, tmp,
                                               M, 768, 512, 512, 512, 768);
  k_ln_relu<<<M, 256, 0, stream>>>(tmp, g_ln2, b_ln2, hhbf, 768, 768, 768);

  // 6) fused mu/logsig GEMMs + Gaussian log-prob accumulation
  k_zero_lp<<<64, 256, 0, stream>>>(lp, M);
  k_musig<<<1536, 256, 0, stream>>>(hhbf, Wmub, Wsigb, b_mu, b_sig, x, lp);

  // 7) masked mean over valid timesteps
  k_final<<<64, 256, 0, stream>>>(lp, lengths, (float*)d_out);
}
